// CrossAttention_25142738551026
// MI455X (gfx1250) — matrix operand order repaired and checked by reading, compiled, not run on hardware
//
#include <hip/hip_runtime.h>
#include <hip/hip_bf16.h>
#include <float.h>
#include <math.h>

// ---------------------------------------------------------------------------
// CrossAttention for MI455X (gfx1250, wave32, WMMA).
// B=4, T=1024, C=1024, H=16, KV=8, D=64, L=32, QK_NORM_SCALE=10.
// All GEMMs run on v_wmma_f32_16x16x32_f16 (f16 operands, f32 accum).
// attn_bias (the dominant 64MB stream) is staged LDS-ward with
// GLOBAL_LOAD_ASYNC_TO_LDS_B128 so it overlaps the QK^T WMMAs.
// ---------------------------------------------------------------------------

constexpr int Bc  = 4;
constexpr int Tc  = 1024;
constexpr int Cc  = 1024;
constexpr int Hc  = 16;
constexpr int KVc = 8;
constexpr int Dc  = 64;
constexpr int Lc  = 32;

typedef __attribute__((ext_vector_type(16))) _Float16 v16h;
typedef __attribute__((ext_vector_type(8)))  float    v8f;

#define WMMA_F16(a, b, c) \
  __builtin_amdgcn_wmma_f32_16x16x32_f16(false, (a), false, (b), (short)0, (c), false, false)

// Async copy: 16 bytes global -> LDS per active lane (tracked by ASYNCcnt).
// VDST = LDS byte address (u32 VGPR), VADDR = 64-bit global address.
__device__ inline void async_g2l_b128(void* lds_ptr, const void* gptr) {
  unsigned lds_off = (unsigned)(size_t)lds_ptr;  // addr[31:0] = LDS offset
  unsigned long long ga = (unsigned long long)(size_t)gptr;
  asm volatile("global_load_async_to_lds_b128 %0, %1, off"
               :: "v"(lds_off), "v"(ga) : "memory");
}
__device__ inline void wait_asynccnt0() {
  asm volatile("s_wait_asynccnt 0x0" ::: "memory");
}

// ---- Fragment builders (ISA 7.12.2 layouts, wave32) -----------------------
// A-matrix 16x32 f16, M x K, row-major source with leading dim `ld`:
//   lane: m = lane&15, half = lane>>4
//   elements 0..7  <-> K =      8*half + 0..7
//   elements 8..15 <-> K = 16 + 8*half + 0..7
__device__ inline v16h load_A_f16(const _Float16* __restrict__ src, int ld) {
  const int lane = threadIdx.x & 31;
  const int m = lane & 15, hf = lane >> 4;
  const _Float16* p = src + (size_t)m * ld + 8 * hf;
  v16h a;
#pragma unroll
  for (int i = 0; i < 8; ++i) { a[i] = p[i]; a[i + 8] = p[i + 16]; }
  return a;
}

__device__ inline v16h load_A_f32(const float* __restrict__ src, int ld) {
  const int lane = threadIdx.x & 31;
  const int m = lane & 15, hf = lane >> 4;
  const float* p = src + (size_t)m * ld + 8 * hf;
  v16h a;
#pragma unroll
  for (int i = 0; i < 8; ++i) {
    a[i]     = (_Float16)p[i];
    a[i + 8] = (_Float16)p[i + 16];
  }
  return a;
}

// B-matrix 32x16 f16, K x N. Source stored TRANSPOSED: src[n*ldk + k] = B[k][n].
//   lane: n = lane&15, half = lane>>4; elements 0..7 <-> K = 8*half + 0..7, 8..15 <-> K = 16 + 8*half + 0..7
__device__ inline v16h load_B_T(const _Float16* __restrict__ src, int ldk) {
  const int lane = threadIdx.x & 31;
  const int n = lane & 15, hf = lane >> 4;
  const _Float16* p = src + (size_t)n * ldk + 8 * hf;
  v16h b;
#pragma unroll
  for (int i = 0; i < 8; ++i) { b[i] = p[i]; b[i + 8] = p[i + 16]; }
  return b;
}

// ---------------------------------------------------------------------------
// Kernel 1: fused QKV projection + l2norm + scale + RoPE.
// grid = (T/16, H + 2*KV, B); block = 1 wave (32).
// Each block computes one 16-token x 64 (= one head's D) tile, K-loop C=1024.
// Q pre-scaled by (D^-0.5 / QK_NORM_SCALE) = 0.0125. V stored transposed
// [b][kv][d][t] so attention V B-fragments are contiguous global loads.
// ---------------------------------------------------------------------------
__global__ __launch_bounds__(32) void qkv_kernel(
    const float* __restrict__ x, const float* __restrict__ enc,
    const float* __restrict__ freqs,
    const float* __restrict__ Wq, const float* __restrict__ Wk,
    const float* __restrict__ Wv,
    const float* __restrict__ q_scale, const float* __restrict__ k_scale,
    _Float16* __restrict__ qws, _Float16* __restrict__ kws,
    _Float16* __restrict__ vtws) {
  __shared__ alignas(32) _Float16 ldsB[64 * 32];  // W chunk transposed [n][k]
  __shared__ alignas(32) float    ldsO[16 * 64];  // epilogue re-layout

  const int lane = threadIdx.x & 31;
  const int t0 = blockIdx.x * 16;
  const int hh = blockIdx.y;
  const int b  = blockIdx.z;

  int kind, head, ldw;
  const float *src, *W;
  if (hh < Hc)            { kind = 0; head = hh;            src = x;   W = Wq; ldw = Hc * Dc;  }
  else if (hh < Hc + KVc) { kind = 1; head = hh - Hc;       src = enc; W = Wk; ldw = KVc * Dc; }
  else                    { kind = 2; head = hh - Hc - KVc; src = enc; W = Wv; ldw = KVc * Dc; }
  const int colbase = head * Dc;

  const float* Arow = src + ((size_t)b * Tc + t0) * Cc;

  v8f acc[4] = {};
  for (int k0 = 0; k0 < Cc; k0 += 32) {
    // Stage W[k0+kk][colbase + n] -> ldsB[n][kk] (f32 -> f16, transposed)
    const float* wrow = W + (size_t)(k0 + lane) * ldw + colbase;
#pragma unroll 8
    for (int n = 0; n < 64; ++n) ldsB[n * 32 + lane] = (_Float16)wrow[n];
    __syncthreads();

    v16h a = load_A_f32(Arow + k0, Cc);
#pragma unroll
    for (int c = 0; c < 4; ++c) {
      v16h bf = load_B_T(ldsB + 16 * c * 32, 32);
      acc[c] = WMMA_F16(a, bf, acc[c]);
    }
    __syncthreads();
  }

  // ---- Epilogue: re-layout 16x64 tile so one lane owns half a row ----
  {
    const int nd = lane & 15, hf = lane >> 4;
#pragma unroll
    for (int c = 0; c < 4; ++c)
#pragma unroll
      for (int r = 0; r < 8; ++r)
        ldsO[(r + 8 * hf) * 64 + 16 * c + nd] = acc[c][r];
  }
  __syncthreads();

  const int m = lane & 15, hf = lane >> 4, c0 = 32 * hf;
  const int t = t0 + m;
  float vals[32];
#pragma unroll
  for (int i = 0; i < 32; ++i) vals[i] = ldsO[m * 64 + c0 + i];

  if (kind != 2) {  // l2norm over D=64 + per-d scale (q/k only)
    float ss = 0.f;
#pragma unroll
    for (int i = 0; i < 32; ++i) ss += vals[i] * vals[i];
    ss += __shfl_xor(ss, 16, 32);
    const float inv = 1.0f / fmaxf(sqrtf(ss), 1e-12f);
    const float* sc = (kind == 0) ? q_scale : k_scale;
#pragma unroll
    for (int i = 0; i < 32; ++i) vals[i] *= inv * sc[c0 + i];
  }
  if (hf == 0) {  // RoPE on d = 0..31 (applied to q, k AND v per reference)
    const float* fr = freqs + (size_t)t * Lc;
#pragma unroll
    for (int i = 0; i < 16; ++i) {
      const float f = fr[2 * i];
      const float cs = cosf(f), sn = sinf(f);
      const float e = vals[2 * i], o = vals[2 * i + 1];
      vals[2 * i]     = e * cs - o * sn;
      vals[2 * i + 1] = o * cs + e * sn;
    }
  }
  if (kind == 0) {
#pragma unroll
    for (int i = 0; i < 32; ++i) vals[i] *= 0.0125f;  // (1/8)/10
    _Float16* dst = qws + (((size_t)b * Hc + head) * Tc + t) * Dc + c0;
#pragma unroll
    for (int i = 0; i < 32; ++i) dst[i] = (_Float16)vals[i];
  } else if (kind == 1) {
    _Float16* dst = kws + (((size_t)b * KVc + head) * Tc + t) * Dc + c0;
#pragma unroll
    for (int i = 0; i < 32; ++i) dst[i] = (_Float16)vals[i];
  } else {  // V stored transposed [b][kv][d][t]
    _Float16* dst = vtws + ((size_t)b * KVc + head) * Dc * Tc;
#pragma unroll
    for (int i = 0; i < 32; ++i) dst[(size_t)(c0 + i) * Tc + t] = (_Float16)vals[i];
  }
}

// ---------------------------------------------------------------------------
// Kernel 2: flash attention with bias + causal mask.
// grid = (T/16, H, B); block = 1 wave. kv head = h & 7 (tile x2 GQA).
// Streams 32-key chunks: bias tile copied global->LDS asynchronously while
// S = Q K^T runs on the matrix pipes; then online softmax; then O += P V.
// ---------------------------------------------------------------------------
__global__ __launch_bounds__(32) void attn_kernel(
    const float* __restrict__ attn_bias,
    const _Float16* __restrict__ qws, const _Float16* __restrict__ kws,
    const _Float16* __restrict__ vtws, _Float16* __restrict__ yws) {
  __shared__ alignas(32) _Float16 ldsP[16 * 32];
  __shared__ alignas(16) float    ldsBias[16 * 32];

  const int lane = threadIdx.x & 31;
  const int t0 = blockIdx.x * 16;
  const int h  = blockIdx.y;
  const int b  = blockIdx.z;
  const int kv = h & (KVc - 1);
  const int nd = lane & 15, hf = lane >> 4;

  const _Float16* qrow  = qws + (((size_t)b * Hc + h) * Tc + t0) * Dc;
  const v16h aq0 = load_A_f16(qrow, Dc);        // K-dim = d 0..31
  const v16h aq1 = load_A_f16(qrow + 32, Dc);   // K-dim = d 32..63
  const _Float16* kbase = kws  + ((size_t)b * KVc + kv) * Tc * Dc;
  const _Float16* vtb   = vtws + ((size_t)b * KVc + kv) * Dc * Tc;
  const float*    biasb = attn_bias + (size_t)h * Tc * Tc;

  float mrun[8], lrun[8];
#pragma unroll
  for (int r = 0; r < 8; ++r) { mrun[r] = -FLT_MAX; lrun[r] = 0.f; }
  v8f o[4] = {};

  const int jmax = (t0 + 15) >> 5;
  for (int j = 0; j <= jmax; ++j) {
    const int jj = j * 32;

    // ---- async-stage the 16x32 f32 bias tile into LDS (ASYNCcnt) --------
    // 2KB = 128 x b128; 32 lanes x 4 issues. row = chunk/8, col = (chunk%8)*4
#pragma unroll
    for (int i = 0; i < 4; ++i) {
      const int e   = i * 32 + lane;
      const int row = e >> 3;
      const int col = (e & 7) * 4;
      async_g2l_b128(&ldsBias[row * 32 + col],
                     biasb + (size_t)(t0 + row) * Tc + jj + col);
    }
    // prefetch next K chunk while we compute (global_prefetch_b8)
    if (j < jmax)
      __builtin_prefetch(kbase + (size_t)(jj + 32) * Dc + lane * 4, 0, 3);

    // ---- S tile (16 x 32): two 16x16 WMMA tiles, K=64 split as 2x32 ----
    v8f s[2];
#pragma unroll
    for (int tl = 0; tl < 2; ++tl) {
      const _Float16* krow = kbase + (size_t)(jj + 16 * tl) * Dc;
      const v16h b0 = load_B_T(krow, Dc);       // K^T: n=key, k=d 0..31
      const v16h b1 = load_B_T(krow + 32, Dc);  // d 32..63
      v8f z = {};
      z = WMMA_F16(aq0, b0, z);
      s[tl] = WMMA_F16(aq1, b1, z);
    }

    wait_asynccnt0();
    __syncthreads();

    // ---- bias + causal mask (D-layout: row = r + 8*half, col = lane&15) ----
    float sv[2][8];
#pragma unroll
    for (int tl = 0; tl < 2; ++tl)
#pragma unroll
      for (int r = 0; r < 8; ++r) {
        const int row = r + 8 * hf;       // t - t0
        const int t   = t0 + row;
        const int key = jj + 16 * tl + nd;
        float v = s[tl][r] * 0.125f;  // default_scale (q already *0.0125)
        v += (key > t) ? (-(FLT_MAX * 0.5f)) : ldsBias[row * 32 + 16 * tl + nd];
        sv[tl][r] = v;
      }

    // ---- online softmax (rows live in 16-lane groups; xor-reduce 1,2,4,8) --
#pragma unroll
    for (int r = 0; r < 8; ++r) {
      float mx = fmaxf(sv[0][r], sv[1][r]);
#pragma unroll
      for (int d = 1; d < 16; d <<= 1) mx = fmaxf(mx, __shfl_xor(mx, d, 32));
      const float mnew = fmaxf(mrun[r], mx);
      const float corr = expf(mrun[r] - mnew);
      const float p0 = expf(sv[0][r] - mnew);
      const float p1 = expf(sv[1][r] - mnew);
      float psum = p0 + p1;
#pragma unroll
      for (int d = 1; d < 16; d <<= 1) psum += __shfl_xor(psum, d, 32);
      lrun[r] = lrun[r] * corr + psum;
      mrun[r] = mnew;
      sv[0][r] = p0; sv[1][r] = p1;
#pragma unroll
      for (int c = 0; c < 4; ++c) o[c][r] *= corr;
    }

    // ---- P -> LDS (row-major 16x32 f16) -> A fragment ----
#pragma unroll
    for (int tl = 0; tl < 2; ++tl)
#pragma unroll
      for (int r = 0; r < 8; ++r)
        ldsP[(r + 8 * hf) * 32 + 16 * tl + nd] = (_Float16)sv[tl][r];
    __syncthreads();
    const v16h ap = load_A_f16(ldsP, 32);

    // ---- O += P @ V  (V transposed: src[n=d][k=key], contiguous keys) ----
#pragma unroll
    for (int c = 0; c < 4; ++c) {
      const v16h bv = load_B_T(vtb + (size_t)16 * c * Tc + jj, Tc);
      o[c] = WMMA_F16(ap, bv, o[c]);
    }
    __syncthreads();  // protect ldsP / ldsBias reuse next iteration
  }

  // ---- normalize + store y as f16 in [b][t][h*D+d] for the Wo GEMM ----
  _Float16* yrow = yws + ((size_t)b * Tc + t0) * (Hc * Dc) + h * Dc;
#pragma unroll
  for (int c = 0; c < 4; ++c)
#pragma unroll
    for (int r = 0; r < 8; ++r) {
      const int m = r + 8 * hf;
      yrow[(size_t)m * (Hc * Dc) + 16 * c + nd] = (_Float16)(o[c][r] / lrun[r]);
    }
}

// ---------------------------------------------------------------------------
// Kernel 3: output projection  (B*T, 1024) x (1024, 1024) -> f32 out.
// grid = (B*T/16, C/64); block = 1 wave.
// ---------------------------------------------------------------------------
__global__ __launch_bounds__(32) void oproj_kernel(
    const _Float16* __restrict__ yws, const float* __restrict__ Wo,
    float* __restrict__ out) {
  __shared__ alignas(32) _Float16 ldsB[64 * 32];

  const int lane = threadIdx.x & 31;
  const int row0 = blockIdx.x * 16;  // flattened over B*T
  const int col0 = blockIdx.y * 64;
  const _Float16* Arow = yws + (size_t)row0 * (Hc * Dc);

  v8f acc[4] = {};
  for (int k0 = 0; k0 < Hc * Dc; k0 += 32) {
    const float* wrow = Wo + (size_t)(k0 + lane) * Cc + col0;
#pragma unroll 8
    for (int n = 0; n < 64; ++n) ldsB[n * 32 + lane] = (_Float16)wrow[n];
    __syncthreads();

    v16h a = load_A_f16(Arow + k0, Hc * Dc);
#pragma unroll
    for (int c = 0; c < 4; ++c) {
      v16h bf = load_B_T(ldsB + 16 * c * 32, 32);
      acc[c] = WMMA_F16(a, bf, acc[c]);
    }
    __syncthreads();
  }

  const int nd = lane & 15, hf = lane >> 4;
#pragma unroll
  for (int c = 0; c < 4; ++c)
#pragma unroll
    for (int r = 0; r < 8; ++r)
      out[(size_t)(row0 + r + 8 * hf) * Cc + col0 + 16 * c + nd] = acc[c][r];
}

// ---------------------------------------------------------------------------
extern "C" void kernel_launch(void* const* d_in, const int* in_sizes, int n_in,
                              void* d_out, int out_size, void* d_ws,
                              size_t ws_size, hipStream_t stream) {
  (void)in_sizes; (void)n_in; (void)out_size; (void)ws_size;
  const float* x     = (const float*)d_in[0];
  const float* enc   = (const float*)d_in[1];
  const float* freqs = (const float*)d_in[2];
  const float* bias  = (const float*)d_in[3];
  const float* Wq    = (const float*)d_in[4];
  const float* Wk    = (const float*)d_in[5];
  const float* Wv    = (const float*)d_in[6];
  const float* Wo    = (const float*)d_in[7];
  const float* qs    = (const float*)d_in[8];
  const float* ks    = (const float*)d_in[9];
  float* out = (float*)d_out;

  // Workspace layout (f16): q 8MB | k 4MB | vT 4MB | y 8MB  (24 MB total)
  char* ws = (char*)d_ws;
  _Float16* qws  = (_Float16*)(ws);
  _Float16* kws  = (_Float16*)(ws + (size_t)8  * 1024 * 1024);
  _Float16* vtws = (_Float16*)(ws + (size_t)12 * 1024 * 1024);
  _Float16* yws  = (_Float16*)(ws + (size_t)16 * 1024 * 1024);

  const dim3 blk(32, 1, 1);
  qkv_kernel<<<dim3(Tc / 16, Hc + 2 * KVc, Bc), blk, 0, stream>>>(
      x, enc, freqs, Wq, Wk, Wv, qs, ks, qws, kws, vtws);
  attn_kernel<<<dim3(Tc / 16, Hc, Bc), blk, 0, stream>>>(
      bias, qws, kws, vtws, yws);
  oproj_kernel<<<dim3((Bc * Tc) / 16, Cc / 64, 1), blk, 0, stream>>>(
      yws, Wo, out);
}
